// SCAGRU_25993142076011
// MI455X (gfx1250) — matrix operand order repaired and checked by reading, not compiled, not run on hardware
//
#include <hip/hip_runtime.h>
#include <hip/hip_bf16.h>

#define T_STEPS  1024
#define BATCH    128
#define INDIM_K  256
#define HID      512
#define NBLOCKS  32
#define NTHREADS 256

typedef __attribute__((ext_vector_type(16))) __bf16 v16bf;
typedef __attribute__((ext_vector_type(8)))  __bf16 v8bf;
typedef __attribute__((ext_vector_type(8)))  float  v8f;

// ---------------------------------------------------------------------------
// Fast activations (v_exp_f32 transcendentals only)
// ---------------------------------------------------------------------------
__device__ __forceinline__ float fast_sigmoid(float x) {
    return 1.0f / (1.0f + __expf(-x));
}
__device__ __forceinline__ float fast_tanh(float x) {
    float e = __expf(-2.0f * x);
    return (1.0f - e) / (1.0f + e);
}

// ---------------------------------------------------------------------------
// Async memory->LDS copy (gfx1250 GLOBAL_LOAD_ASYNC_TO_LDS_B128, ASYNCcnt).
// No VGPR round-trip; LDS byte address = low 32 bits of the generic pointer
// (ISA 10.2: LDS aperture addresses truncate to their low 32 bits).
// ---------------------------------------------------------------------------
__device__ __forceinline__ void async_copy_b128(void* lds_dst, const void* gsrc) {
    unsigned           ldsoff = (unsigned)(unsigned long long)lds_dst;
    unsigned long long gaddr  = (unsigned long long)gsrc;
    asm volatile("global_load_async_to_lds_b128 %0, %1, off"
                 :: "v"(ldsoff), "v"(gaddr) : "memory");
}
__device__ __forceinline__ void wait_async_all() {
    asm volatile("s_wait_asynccnt 0" ::: "memory");
}

// Copy one contiguous 16-row bf16 tile (nbytes = 16*ncols*2) into LDS, async.
__device__ __forceinline__ void stage_async_bf16(__bf16* lds, const __bf16* __restrict__ src,
                                                 int nelem, int tid) {
    const int chunks = nelem >> 3;                    // 16B per chunk
    for (int c = tid; c < chunks; c += NTHREADS)
        async_copy_b128(lds + (c << 3), src + (c << 3));
}

// Synchronous f32 -> bf16 stage (conversion must go through VGPRs anyway).
__device__ __forceinline__ void stage_f32_tile(__bf16* dst, const float* __restrict__ src,
                                               int nelem, int tid) {
    const int chunks = nelem >> 3;
    for (int c = tid; c < chunks; c += NTHREADS) {
        int k = c << 3;
        v8f f = *(const v8f*)(src + k);
        v8bf b;
#pragma unroll
        for (int i = 0; i < 8; ++i) b[i] = (__bf16)f[i];
        *(v8bf*)(dst + k) = b;
    }
}

// ---------------------------------------------------------------------------
// WMMA fragment loaders (contiguous-k order, same for A and B, wave32)
// A 16x32 bf16: lane L -> row = L&15 ; 16 contiguous K[kb..kb+15]
//               kb = (L>>4)*16.  Source is the block-shared LDS tile.
// ---------------------------------------------------------------------------
__device__ __forceinline__ v16bf load_a_lds(const __bf16* L, int ld, int k0, int lane) {
    int r  = lane & 15;
    int kb = k0 + ((lane >> 4) << 4);
    const __bf16* p = L + r * ld + kb;
    v8bf lo = *(const v8bf*)p;            // ds_load_b128
    v8bf hi = *(const v8bf*)(p + 8);      // ds_load_b128
    v16bf a;
#pragma unroll
    for (int i = 0; i < 8; ++i) { a[i] = lo[i]; a[8 + i] = hi[i]; }
    return a;
}

// B 32x16 bf16: lane L -> col n = L&15 ; 16 contiguous K at kb = (L>>4)*16.
// Weights pre-transposed to Wt[n][k] so this is one 32B contiguous load (L2-resident).
__device__ __forceinline__ v16bf load_b(const __bf16* __restrict__ Wt, int ldk,
                                        int n0, int k0, int lane) {
    int n  = n0 + (lane & 15);
    int kb = k0 + ((lane >> 4) << 4);
    return *(const v16bf*)(Wt + n * ldk + kb);
}

__device__ __forceinline__ v8f wmma_bf16(v16bf a, v16bf b, v8f c) {
    return __builtin_amdgcn_wmma_f32_16x16x32_bf16(false, a, false, b,
                                                   (short)0, c, false, false);
}

// ---------------------------------------------------------------------------
// Grid-wide barrier (all NBLOCKS blocks resident; monotonic phase)
// ---------------------------------------------------------------------------
__device__ __forceinline__ void grid_barrier(unsigned* cnt, unsigned* phase,
                                             unsigned nblk, unsigned& barphase) {
    __syncthreads();
    if (threadIdx.x == 0) {
        __threadfence();
        unsigned target = barphase + 1;
        unsigned a = atomicAdd(cnt, 1u);
        if (a == nblk - 1) {
            atomicExch(cnt, 0u);
            __threadfence();
            atomicAdd(phase, 1u);
        } else {
            while (atomicAdd(phase, 0u) < target) { __builtin_amdgcn_s_sleep(2); }
        }
        __threadfence();
    }
    barphase += 1;
    __syncthreads();
}

// ---------------------------------------------------------------------------
// Prep: transposed bf16 weights. Wt[n*K + k] = W[k*N + n]   (N==HID always)
// ---------------------------------------------------------------------------
__global__ void wprep_kernel(const float* __restrict__ W, __bf16* __restrict__ Wt,
                             int K, int N) {
    int idx = blockIdx.x * blockDim.x + threadIdx.x;
    if (idx >= K * N) return;
    int n = idx / K;
    int k = idx - n * K;
    Wt[idx] = (__bf16)W[k * N + n];
}

__global__ void sync_init_kernel(unsigned* sync) {
    if (threadIdx.x < 2) sync[threadIdx.x] = 0u;
}

// ---------------------------------------------------------------------------
// Persistent recurrent kernel: 32 blocks x 256 threads = 256 waves.
// All 8 waves of a block share one batch-row block (m0): the A tile is staged
// once per block into LDS (async bf16 DMA when the source is already bf16).
// Wave w owns output tile (m = w>>5, n = w&31).
// ---------------------------------------------------------------------------
__global__ __launch_bounds__(NTHREADS) void scagru_persistent(
    const float* __restrict__ input,   // [T, B, INDIM]
    const float* __restrict__ state,   // [B, H]
    const float* __restrict__ b_z, const float* __restrict__ b_r,
    const float* __restrict__ b_h, const float* __restrict__ b_c,
    const __bf16* __restrict__ Wt_zh, const __bf16* __restrict__ Wt_rh,
    const __bf16* __restrict__ Wt_hh, const __bf16* __restrict__ Wt_ch,
    const __bf16* __restrict__ Wt_zx, const __bf16* __restrict__ Wt_rx,
    const __bf16* __restrict__ Wt_hx,
    float* __restrict__ h,             // [B, H] f32 master state
    __bf16* __restrict__ hbf,          // [B, H] bf16 mirror of h
    float* __restrict__ zbuf, float* __restrict__ xhbuf,
    float* __restrict__ cbuf,
    __bf16* __restrict__ hrbuf,        // [B, H] bf16 h*r
    unsigned* __restrict__ sync,       // [0]=count [1]=phase
    float* __restrict__ out)
{
    __shared__ __bf16 lds_a[16 * HID];      // h tile (phase 1) / hr tile (phase 2)
    __shared__ __bf16 lds_x[16 * INDIM_K];  // x_t tile (phase 1)

    const int tid  = threadIdx.x;
    const int lane = tid & 31;
    const int wid  = (blockIdx.x * NTHREADS + tid) >> 5;  // 0..255
    const int m0 = (wid >> 5) << 4;    // identical for all waves in the block
    const int n0 = (wid & 31) << 4;
    unsigned* cnt   = sync;
    unsigned* phase = sync + 1;
    unsigned barphase = 0;

    // lane -> C/D coordinates (ISA 7.12.2: N = lane&15, M = vgpr + 8*(lane>>4))
    const int colN    = n0 + (lane & 15);
    const int rowBase = m0 + ((lane >> 4) << 3);

    // ---- context gate: c = relu(state @ W_ch + b_c); h = state ----
    {
        stage_f32_tile(lds_a, state + m0 * HID, 16 * HID, tid);
        __syncthreads();
        v8f acc = {};
        for (int k0 = 0; k0 < HID; k0 += 32) {
            v16bf a = load_a_lds(lds_a, HID, k0, lane);
            acc = wmma_bf16(a, load_b(Wt_ch, HID, n0, k0, lane), acc);
        }
        float bc = b_c[colN];
#pragma unroll
        for (int v = 0; v < 8; ++v) {
            int idx = (rowBase + v) * HID + colN;
            float cv = acc[v] + bc;
            cbuf[idx] = cv > 0.0f ? cv : 0.0f;
            float sv = state[idx];
            h[idx]   = sv;
            hbf[idx] = (__bf16)sv;
        }
    }
    grid_barrier(cnt, phase, NBLOCKS, barphase);

    for (int t = 0; t < T_STEPS; ++t) {
        const float* xt = input + (size_t)t * (BATCH * INDIM_K);

        // speculative prefetch of next timestep's input (global_prefetch_b8)
        if (t + 1 < T_STEPS) {
            const float* xn = xt + BATCH * INDIM_K;
            __builtin_prefetch(xn + ((size_t)(blockIdx.x * NTHREADS + tid) << 4), 0, 1);
        }

        // -------- phase 1: z, r gates + x-projection of candidate ------------
        stage_async_bf16(lds_a, hbf + m0 * HID, 16 * HID, tid);   // async DMA
        stage_f32_tile(lds_x, xt + m0 * INDIM_K, 16 * INDIM_K, tid);
        wait_async_all();
        __syncthreads();

        v8f az = {}, ar = {}, axh = {};
        for (int k0 = 0; k0 < HID; k0 += 32) {
            v16bf a = load_a_lds(lds_a, HID, k0, lane);
            az = wmma_bf16(a, load_b(Wt_zh, HID, n0, k0, lane), az);
            ar = wmma_bf16(a, load_b(Wt_rh, HID, n0, k0, lane), ar);
        }
        for (int k0 = 0; k0 < INDIM_K; k0 += 32) {
            v16bf a = load_a_lds(lds_x, INDIM_K, k0, lane);
            az  = wmma_bf16(a, load_b(Wt_zx, INDIM_K, n0, k0, lane), az);
            ar  = wmma_bf16(a, load_b(Wt_rx, INDIM_K, n0, k0, lane), ar);
            axh = wmma_bf16(a, load_b(Wt_hx, INDIM_K, n0, k0, lane), axh);
        }
        {
            float bz = b_z[colN], br = b_r[colN], bh = b_h[colN];
#pragma unroll
            for (int v = 0; v < 8; ++v) {
                int idx = (rowBase + v) * HID + colN;
                float cv = cbuf[idx];
                float zz = fast_sigmoid(az[v] + bz + cv);
                float rr = fast_sigmoid(ar[v] + br + cv);
                zbuf[idx]  = zz;
                xhbuf[idx] = axh[v] + bh + cv;
                hrbuf[idx] = (__bf16)(h[idx] * rr);
            }
        }
        grid_barrier(cnt, phase, NBLOCKS, barphase);

        // -------- phase 2: candidate GEMM + gated update ----------------------
        stage_async_bf16(lds_a, hrbuf + m0 * HID, 16 * HID, tid); // async DMA
        wait_async_all();
        __syncthreads();

        v8f ah = {};
        for (int k0 = 0; k0 < HID; k0 += 32) {
            v16bf a = load_a_lds(lds_a, HID, k0, lane);
            ah = wmma_bf16(a, load_b(Wt_hh, HID, n0, k0, lane), ah);
        }
#pragma unroll
        for (int v = 0; v < 8; ++v) {
            int idx = (rowBase + v) * HID + colN;
            float ht = fast_tanh(ah[v] + xhbuf[idx]);
            float zz = zbuf[idx];
            float hv = h[idx];
            float hn = (1.0f - zz) * hv + zz * ht;
            h[idx]   = hn;
            hbf[idx] = (__bf16)hn;
        }
        grid_barrier(cnt, phase, NBLOCKS, barphase);
    }

    // final state -> output
    for (int i = blockIdx.x * NTHREADS + tid; i < BATCH * HID; i += NBLOCKS * NTHREADS)
        out[i] = h[i];
}

// ---------------------------------------------------------------------------
// Host launch
// ---------------------------------------------------------------------------
extern "C" void kernel_launch(void* const* d_in, const int* in_sizes, int n_in,
                              void* d_out, int out_size, void* d_ws, size_t ws_size,
                              hipStream_t stream) {
    const float* input = (const float*)d_in[0];
    const float* state = (const float*)d_in[1];
    const float* W_zh  = (const float*)d_in[2];
    const float* W_zx  = (const float*)d_in[3];
    const float* b_z   = (const float*)d_in[4];
    const float* W_rh  = (const float*)d_in[5];
    const float* W_rx  = (const float*)d_in[6];
    const float* b_r   = (const float*)d_in[7];
    const float* W_hh  = (const float*)d_in[8];
    const float* W_hx  = (const float*)d_in[9];
    const float* b_h   = (const float*)d_in[10];
    const float* W_ch  = (const float*)d_in[11];
    const float* b_c   = (const float*)d_in[12];
    float* out = (float*)d_out;

    // ---- workspace carving (all offsets multiples of 256 B) ----
    char* ws = (char*)d_ws;
    size_t o = 0;
    const size_t BH_F = (size_t)BATCH * HID * sizeof(float);    // 256 KB
    const size_t BH_B = (size_t)BATCH * HID * sizeof(__bf16);   // 128 KB
    const size_t HH_B = (size_t)HID * HID * sizeof(__bf16);     // 512 KB
    const size_t IH_B = (size_t)INDIM_K * HID * sizeof(__bf16); // 256 KB

    float*  h     = (float*)(ws + o);  o += BH_F;
    float*  zbuf  = (float*)(ws + o);  o += BH_F;
    float*  xhbuf = (float*)(ws + o);  o += BH_F;
    float*  cbuf  = (float*)(ws + o);  o += BH_F;
    __bf16* hbf   = (__bf16*)(ws + o); o += BH_B;
    __bf16* hrbuf = (__bf16*)(ws + o); o += BH_B;
    __bf16* Wt_zh = (__bf16*)(ws + o); o += HH_B;
    __bf16* Wt_rh = (__bf16*)(ws + o); o += HH_B;
    __bf16* Wt_hh = (__bf16*)(ws + o); o += HH_B;
    __bf16* Wt_ch = (__bf16*)(ws + o); o += HH_B;
    __bf16* Wt_zx = (__bf16*)(ws + o); o += IH_B;
    __bf16* Wt_rx = (__bf16*)(ws + o); o += IH_B;
    __bf16* Wt_hx = (__bf16*)(ws + o); o += IH_B;
    unsigned* sync = (unsigned*)(ws + o); o += 256;
    (void)ws_size; (void)n_in; (void)in_sizes; (void)out_size;

    // ---- weight prep: transpose + convert to bf16 ----
    {
        const int HH = HID * HID, IH = INDIM_K * HID, TPB = 256;
        wprep_kernel<<<(HH + TPB - 1) / TPB, TPB, 0, stream>>>(W_zh, Wt_zh, HID, HID);
        wprep_kernel<<<(HH + TPB - 1) / TPB, TPB, 0, stream>>>(W_rh, Wt_rh, HID, HID);
        wprep_kernel<<<(HH + TPB - 1) / TPB, TPB, 0, stream>>>(W_hh, Wt_hh, HID, HID);
        wprep_kernel<<<(HH + TPB - 1) / TPB, TPB, 0, stream>>>(W_ch, Wt_ch, HID, HID);
        wprep_kernel<<<(IH + TPB - 1) / TPB, TPB, 0, stream>>>(W_zx, Wt_zx, INDIM_K, HID);
        wprep_kernel<<<(IH + TPB - 1) / TPB, TPB, 0, stream>>>(W_rx, Wt_rx, INDIM_K, HID);
        wprep_kernel<<<(IH + TPB - 1) / TPB, TPB, 0, stream>>>(W_hx, Wt_hx, INDIM_K, HID);
    }
    sync_init_kernel<<<1, 64, 0, stream>>>(sync);

    scagru_persistent<<<NBLOCKS, NTHREADS, 0, stream>>>(
        input, state, b_z, b_r, b_h, b_c,
        Wt_zh, Wt_rh, Wt_hh, Wt_ch, Wt_zx, Wt_rx, Wt_hx,
        h, hbf, zbuf, xhbuf, cbuf, hrbuf, sync, out);
}
